// DeConvRFNet_32641751449621
// MI455X (gfx1250) — matrix operand order repaired and checked by reading, not compiled, not run on hardware
//
#include <hip/hip_runtime.h>
#include <hip/hip_bf16.h>
#include <math.h>

typedef __attribute__((ext_vector_type(16))) __bf16 v16bf;
typedef __attribute__((ext_vector_type(8)))  float  v8f;

#define BN_EPS 1e-5f
#define TB 16            // batch
#define GBM 64           // GEMM M tile
#define GBN 128          // GEMM N tile

__device__ __forceinline__ unsigned short f2bf(float f) {
  unsigned int u = __float_as_uint(f);
  u += 0x7FFFu + ((u >> 16) & 1u);       // round-to-nearest-even to bf16
  return (unsigned short)(u >> 16);
}

// ---------------- NCHW -> CBHW permute ----------------
__global__ void k_nchw_to_cbhw(const float* __restrict__ in, float* __restrict__ out,
                               int C, int HW) {
  long long i = (long long)blockIdx.x * blockDim.x + threadIdx.x;
  long long total = (long long)TB * C * HW;
  if (i >= total) return;
  int s = (int)(i % HW);
  int c = (int)((i / HW) % C);
  int b = (int)(i / ((long long)HW * C));
  out[((long long)c * TB + b) * HW + s] = in[i];
}

// ---------------- direct conv (CBHW), mode 0 = +bias, mode 1 = 2*sigmoid(+bias) ----------------
__global__ void k_direct_conv(const float* __restrict__ x, const float* __restrict__ w,
                              const float* __restrict__ bias, float* __restrict__ y,
                              int Cin, int H, int W, int Cout,
                              int k, int stride, int pad, int Ho, int Wo, int mode) {
  long long i = (long long)blockIdx.x * blockDim.x + threadIdx.x;
  long long total = (long long)Cout * TB * Ho * Wo;
  if (i >= total) return;
  int wo = (int)(i % Wo);
  int ho = (int)((i / Wo) % Ho);
  int b  = (int)((i / ((long long)Wo * Ho)) % TB);
  int co = (int)(i / ((long long)Wo * Ho * TB));
  float acc = bias[co];
  const long long HWi = (long long)H * W;
  for (int ci = 0; ci < Cin; ++ci) {
    const float* xp = x + ((long long)ci * TB + b) * HWi;
    const float* wp = w + ((long long)co * Cin + ci) * k * k;
    for (int kh = 0; kh < k; ++kh) {
      int iy = ho * stride - pad + kh;
      if (iy < 0 || iy >= H) continue;
      for (int kw = 0; kw < k; ++kw) {
        int ix = wo * stride - pad + kw;
        if (ix < 0 || ix >= W) continue;
        acc += xp[(long long)iy * W + ix] * wp[kh * k + kw];
      }
    }
  }
  if (mode == 1) acc = 2.0f / (1.0f + __expf(-acc));
  y[i] = acc;
}

// ---- tiled deformable column builder: cols element (r,n) at ((r/32)*N + n)*32 + (r%32) ----
// one thread per (k-block q, n): writes one full 64B row (4 x b128), zero-fills r >= Krows.
__global__ void k_deform_cols_t(const float* __restrict__ x, const float* __restrict__ off,
                                const float* __restrict__ mod, unsigned short* __restrict__ cols,
                                int Cin, int H, int W, int k, int stride, int pad,
                                int Ho, int Wo, int Krows, int Kpad) {
  const int K = k * k;
  const long long BHW = (long long)TB * Ho * Wo;
  const int Q = Kpad >> 5;
  long long i = (long long)blockIdx.x * blockDim.x + threadIdx.x;
  long long total = (long long)Q * BHW;
  if (i >= total) return;
  long long n = i % BHW;
  int q = (int)(i / BHW);
  int wo = (int)(n % Wo);
  int ho = (int)((n / Wo) % Ho);
  int b  = (int)(n / ((long long)Wo * Ho));
  const long long HWi = (long long)H * W;
  __align__(16) unsigned short tmp[32];
  const int r0 = q * 32;
  int ci = r0 / K;
  int kk = r0 - ci * K;
#pragma unroll
  for (int j = 0; j < 32; ++j) {
    float v = 0.0f;
    if (r0 + j < Krows) {
      float dy = off[(long long)(2 * kk)     * BHW + n];
      float dx = off[(long long)(2 * kk + 1) * BHW + n];
      float mv = mod[(long long)kk * BHW + n];
      int ky = kk / k, kx = kk - ky * k;
      float py = (float)(ho * stride - pad + ky) + dy;
      float px = (float)(wo * stride - pad + kx) + dx;
      float y0 = floorf(py), x0 = floorf(px);
      float wy1 = py - y0, wx1 = px - x0;
      int iy0 = (int)y0, ix0 = (int)x0;
      bool vy0 = (iy0 >= 0) && (iy0 <= H - 1);
      bool vy1 = (iy0 + 1 >= 0) && (iy0 + 1 <= H - 1);
      bool vx0 = (ix0 >= 0) && (ix0 <= W - 1);
      bool vx1 = (ix0 + 1 >= 0) && (ix0 + 1 <= W - 1);
      int cy0 = min(max(iy0, 0), H - 1), cy1 = min(max(iy0 + 1, 0), H - 1);
      int cx0 = min(max(ix0, 0), W - 1), cx1 = min(max(ix0 + 1, 0), W - 1);
      float f00 = (vy0 && vx0) ? (1.0f - wy1) * (1.0f - wx1) : 0.0f;
      float f01 = (vy0 && vx1) ? (1.0f - wy1) * wx1          : 0.0f;
      float f10 = (vy1 && vx0) ? wy1 * (1.0f - wx1)          : 0.0f;
      float f11 = (vy1 && vx1) ? wy1 * wx1                   : 0.0f;
      const float* xp = x + ((long long)ci * TB + b) * HWi;
      v = xp[(long long)cy0 * W + cx0] * f00 + xp[(long long)cy0 * W + cx1] * f01
        + xp[(long long)cy1 * W + cx0] * f10 + xp[(long long)cy1 * W + cx1] * f11;
      v *= mv;
    }
    tmp[j] = f2bf(v);
    if (++kk == K) { kk = 0; ++ci; }
  }
  uint4* dst = (uint4*)(cols + ((long long)q * BHW + n) * 32);
  const uint4* s = (const uint4*)tmp;
  dst[0] = s[0]; dst[1] = s[1]; dst[2] = s[2]; dst[3] = s[3];
}

// ---------------- tiled im2col: same layout, zero-fills padding rows ----------------
__global__ void k_im2col_t(const float* __restrict__ x, unsigned short* __restrict__ cols,
                           int Cin, int H, int W, int k, int stride, int pad,
                           int Ho, int Wo, int Krows, int Kpad) {
  const int K = k * k;
  const long long BHW = (long long)TB * Ho * Wo;
  const int Q = Kpad >> 5;
  long long i = (long long)blockIdx.x * blockDim.x + threadIdx.x;
  long long total = (long long)Q * BHW;
  if (i >= total) return;
  long long n = i % BHW;
  int q = (int)(i / BHW);
  int wo = (int)(n % Wo);
  int ho = (int)((n / Wo) % Ho);
  int b  = (int)(n / ((long long)Wo * Ho));
  const long long HWi = (long long)H * W;
  __align__(16) unsigned short tmp[32];
  const int r0 = q * 32;
  int ci = r0 / K;
  int kk = r0 - ci * K;
#pragma unroll
  for (int j = 0; j < 32; ++j) {
    float v = 0.0f;
    if (r0 + j < Krows) {
      int ky = kk / k, kx = kk - ky * k;
      int iy = ho * stride - pad + ky;
      int ix = wo * stride - pad + kx;
      if (iy >= 0 && iy < H && ix >= 0 && ix < W)
        v = x[((long long)ci * TB + b) * HWi + (long long)iy * W + ix];
    }
    tmp[j] = f2bf(v);
    if (++kk == K) { kk = 0; ++ci; }
  }
  uint4* dst = (uint4*)(cols + ((long long)q * BHW + n) * 32);
  const uint4* s = (const uint4*)tmp;
  dst[0] = s[0]; dst[1] = s[1]; dst[2] = s[2]; dst[3] = s[3];
}

// ---------------- weight pack: fp32 [M][Krows] -> bf16 [MpadT][Kpad] (zero padded) ------------
__global__ void k_pack_w(const float* __restrict__ w, unsigned short* __restrict__ out,
                         int M, int Krows, int MpadT, int Kpad) {
  long long i = (long long)blockIdx.x * blockDim.x + threadIdx.x;
  long long total = (long long)MpadT * Kpad;
  if (i >= total) return;
  int kp = (int)(i % Kpad);
  int mp = (int)(i / Kpad);
  float v = (mp < M && kp < Krows) ? w[(long long)mp * Krows + kp] : 0.0f;
  out[i] = f2bf(v);
}

// ---------------- WMMA GEMM, LDS-free ----------------
// C[M][N] = A[MpadT][Kpad](bf16,row-major) * Bt(bf16, K-tiled: (k,n) -> ((k/32)*N+n)*32+(k%32)) + bias
// 256 threads = 8 wave32 (4 in M x 2 in N); wave = 16M x 64N; fragments load straight from
// global as contiguous b128 pairs (wave covers a contiguous 4KB span of Bt per k-block).
__global__ __launch_bounds__(256) void k_wmma_gemm(
    const unsigned short* __restrict__ A, const unsigned short* __restrict__ Bt,
    const float* __restrict__ bias, float* __restrict__ C,
    int M, int N, int Kpad) {
  const int lane  = threadIdx.x & 31;
  const int wave  = threadIdx.x >> 5;
  const int waveM = wave & 3;
  const int waveN = wave >> 2;
  const long long m0 = (long long)blockIdx.y * GBM;
  const long long n0 = (long long)blockIdx.x * GBN;
  const int Q = Kpad >> 5;

  // A fragment: lane holds row m, k {kbA..kbA+15}
  const int mfrag = (lane & 15) + waveM * 16;
  const int kbA = (lane < 16) ? 0 : 16;
  const int kbB = (lane < 16) ? 0 : 16;
  const unsigned short* aptr = A + (m0 + mfrag) * (long long)Kpad + kbA;
  const unsigned short* bptr0;
  const unsigned short* bptr1;
  const unsigned short* bptr2;
  const unsigned short* bptr3;
  {
    long long nb = n0 + waveN * 64 + (lane & 15);
    bptr0 = Bt + (nb +  0) * 32 + kbB;
    bptr1 = Bt + (nb + 16) * 32 + kbB;
    bptr2 = Bt + (nb + 32) * 32 + kbB;
    bptr3 = Bt + (nb + 48) * 32 + kbB;
  }
  const long long bstep = (long long)N * 32;  // elements per k-block of Bt

  const v8f vz = {0.f, 0.f, 0.f, 0.f, 0.f, 0.f, 0.f, 0.f};
  v8f acc0 = vz, acc1 = vz, acc2 = vz, acc3 = vz;

  for (int q = 0; q < Q; ++q) {
    union { v16bf v; uint4 u4[2]; } af;
    af.u4[0] = *(const uint4*)(aptr);
    af.u4[1] = *(const uint4*)(aptr + 8);
    if (q + 1 < Q) {   // hint next k-block toward caches (global_prefetch_b8)
      __builtin_prefetch((const void*)(aptr + 32), 0, 1);
      __builtin_prefetch((const void*)(bptr0 + bstep), 0, 1);
      __builtin_prefetch((const void*)(bptr2 + bstep), 0, 1);
    }
    union { v16bf v; uint4 u4[2]; } bf0, bf1, bf2, bf3;
    bf0.u4[0] = *(const uint4*)(bptr0); bf0.u4[1] = *(const uint4*)(bptr0 + 8);
    bf1.u4[0] = *(const uint4*)(bptr1); bf1.u4[1] = *(const uint4*)(bptr1 + 8);
    bf2.u4[0] = *(const uint4*)(bptr2); bf2.u4[1] = *(const uint4*)(bptr2 + 8);
    bf3.u4[0] = *(const uint4*)(bptr3); bf3.u4[1] = *(const uint4*)(bptr3 + 8);
    acc0 = __builtin_amdgcn_wmma_f32_16x16x32_bf16(false, af.v, false, bf0.v, (short)0, acc0, false, false);
    acc1 = __builtin_amdgcn_wmma_f32_16x16x32_bf16(false, af.v, false, bf1.v, (short)0, acc1, false, false);
    acc2 = __builtin_amdgcn_wmma_f32_16x16x32_bf16(false, af.v, false, bf2.v, (short)0, acc2, false, false);
    acc3 = __builtin_amdgcn_wmma_f32_16x16x32_bf16(false, af.v, false, bf3.v, (short)0, acc3, false, false);
    aptr += 32;
    bptr0 += bstep; bptr1 += bstep; bptr2 += bstep; bptr3 += bstep;
  }

  // C/D layout: VGPR i holds M = i + 8*(lane>=16), N = lane&15
  const int mrow = waveM * 16 + ((lane < 16) ? 0 : 8);
  const int ncol = waveN * 64 + (lane & 15);
#pragma unroll
  for (int i = 0; i < 8; ++i) {
    const long long m = m0 + mrow + i;
    if (m < M) {
      float bsum = bias[m];
      float* crow = C + m * (long long)N + n0 + ncol;
      crow[ 0] = acc0[i] + bsum;
      crow[16] = acc1[i] + bsum;
      crow[32] = acc2[i] + bsum;
      crow[48] = acc3[i] + bsum;
    }
  }
}

// ---------------- BN stats: one block per channel (deterministic, no fp atomics) ------------
__global__ void k_bn_stats(const float* __restrict__ a, float* __restrict__ mean,
                           float* __restrict__ rstd, long long N) {
  const int c = blockIdx.x;
  const float* p = a + (long long)c * N;
  float s = 0.f, ss = 0.f;
  for (long long i = threadIdx.x; i < N; i += blockDim.x) {
    float v = p[i];
    s += v; ss += v * v;
  }
  __shared__ float sh1[256], sh2[256];
  sh1[threadIdx.x] = s; sh2[threadIdx.x] = ss;
  __syncthreads();
  for (int st = 128; st > 0; st >>= 1) {
    if (threadIdx.x < st) { sh1[threadIdx.x] += sh1[threadIdx.x + st];
                            sh2[threadIdx.x] += sh2[threadIdx.x + st]; }
    __syncthreads();
  }
  if (threadIdx.x == 0) {
    float m = sh1[0] / (float)N;
    float v = sh2[0] / (float)N - m * m;
    mean[c] = m;
    rstd[c] = rsqrtf(v + BN_EPS);
  }
}

__global__ void k_bn_apply(const float* __restrict__ a, const float* __restrict__ mean,
                           const float* __restrict__ rstd, const float* __restrict__ gamma,
                           const float* __restrict__ beta, const float* __restrict__ add,
                           float* __restrict__ out, long long N, int C, int relu) {
  long long i = (long long)blockIdx.x * blockDim.x + threadIdx.x;
  long long total = (long long)C * N;
  if (i >= total) return;
  int c = (int)(i / N);
  float y = (a[i] - mean[c]) * rstd[c] * gamma[c] + beta[c];
  if (add) y += add[i];
  if (relu) y = fmaxf(y, 0.0f);
  out[i] = y;
}

// ================= host orchestration =================
#define L1D(kern, total, ...) do {                                   \
    long long _t = (total);                                          \
    unsigned _g = (unsigned)((_t + 255) / 256);                      \
    kern<<<_g, 256, 0, stream>>>(__VA_ARGS__);                       \
  } while (0)

static void gemm_bn(const float* w, const float* bias, const unsigned short* cols,
                    int Cout, int Krows, int Kpad, int N,
                    unsigned short* wp, float* bufA, float* meanb, float* rstdb,
                    const float* gamma, const float* beta, const float* add,
                    float* out, int relu, hipStream_t stream) {
  int MpadT = (Cout + 63) & ~63;
  L1D(k_pack_w, (long long)MpadT * Kpad, w, wp, Cout, Krows, MpadT, Kpad);
  dim3 g((unsigned)(N / GBN), (unsigned)(MpadT / GBM));
  k_wmma_gemm<<<g, 256, 0, stream>>>(wp, cols, bias, bufA, Cout, N, Kpad);
  k_bn_stats<<<Cout, 256, 0, stream>>>(bufA, meanb, rstdb, (long long)N);
  L1D(k_bn_apply, (long long)Cout * N, bufA, meanb, rstdb, gamma, beta, add, out,
      (long long)N, Cout, relu);
}

static void deform_layer(const float* x, int Cin, int H, int W, int Cout, int k,
                         int stride, int pad,
                         const float* w, const float* b, const float* w_off,
                         const float* b_off, const float* w_mod, const float* b_mod,
                         const float* gamma, const float* beta,
                         float* offb, float* modb, unsigned short* cols,
                         unsigned short* wp, float* bufA, float* meanb, float* rstdb,
                         float* out, hipStream_t stream) {
  int Ho = (H + 2 * pad - k) / stride + 1;
  int Wo = (W + 2 * pad - k) / stride + 1;
  int K = k * k;
  long long N = (long long)TB * Ho * Wo;
  int Krows = Cin * K;
  int Kpad = (Krows + 31) & ~31;
  L1D(k_direct_conv, (long long)(2 * K) * N,
      x, w_off, b_off, offb, Cin, H, W, 2 * K, k, stride, pad, Ho, Wo, 0);
  L1D(k_direct_conv, (long long)K * N,
      x, w_mod, b_mod, modb, Cin, H, W, K, k, stride, pad, Ho, Wo, 1);
  L1D(k_deform_cols_t, (long long)(Kpad >> 5) * N,
      x, offb, modb, cols, Cin, H, W, k, stride, pad, Ho, Wo, Krows, Kpad);
  gemm_bn(w, b, cols, Cout, Krows, Kpad, (int)N, wp, bufA, meanb, rstdb,
          gamma, beta, nullptr, out, 1, stream);
}

extern "C" void kernel_launch(void* const* d_in, const int* in_sizes, int n_in,
                              void* d_out, int out_size, void* d_ws, size_t ws_size,
                              hipStream_t stream) {
  (void)in_sizes; (void)n_in; (void)out_size; (void)ws_size;
  const float* x_in = (const float*)d_in[0];
  const float *d1w = (const float*)d_in[1],  *d1b = (const float*)d_in[2];
  const float *d1wo = (const float*)d_in[3], *d1bo = (const float*)d_in[4];
  const float *d1wm = (const float*)d_in[5], *d1bm = (const float*)d_in[6];
  const float *g1 = (const float*)d_in[7],   *bt1 = (const float*)d_in[8];
  const float *d2w = (const float*)d_in[9],  *d2b = (const float*)d_in[10];
  const float *d2wo = (const float*)d_in[11],*d2bo = (const float*)d_in[12];
  const float *d2wm = (const float*)d_in[13],*d2bm = (const float*)d_in[14];
  const float *g2 = (const float*)d_in[15],  *bt2 = (const float*)d_in[16];
  const float *d3w = (const float*)d_in[17], *d3b = (const float*)d_in[18];
  const float *d3wo = (const float*)d_in[19],*d3bo = (const float*)d_in[20];
  const float *d3wm = (const float*)d_in[21],*d3bm = (const float*)d_in[22];
  const float *g3 = (const float*)d_in[23],  *bt3 = (const float*)d_in[24];
  const float *rwa = (const float*)d_in[25], *rba = (const float*)d_in[26];
  const float *rga = (const float*)d_in[27], *rbta = (const float*)d_in[28];
  const float *rwb = (const float*)d_in[29], *rbb = (const float*)d_in[30];
  const float *rgb = (const float*)d_in[31], *rbtb = (const float*)d_in[32];
  const float *d4w = (const float*)d_in[33], *d4b = (const float*)d_in[34];
  const float *d4wo = (const float*)d_in[35],*d4bo = (const float*)d_in[36];
  const float *d4wm = (const float*)d_in[37],*d4bm = (const float*)d_in[38];
  const float *g4 = (const float*)d_in[39],  *bt4 = (const float*)d_in[40];
  const float *d5w = (const float*)d_in[41], *d5b = (const float*)d_in[42];
  const float *d5wo = (const float*)d_in[43],*d5bo = (const float*)d_in[44];
  const float *d5wm = (const float*)d_in[45],*d5bm = (const float*)d_in[46];
  const float *g5 = (const float*)d_in[47],  *bt5 = (const float*)d_in[48];

  const long long N1 = (long long)TB * 128 * 128;   // 262144
  const long long N2 = (long long)TB * 64 * 64;     // 65536

  char* ws = (char*)d_ws;
  size_t off = 0;
  auto alloc = [&](size_t bytes) -> char* {
    off = (off + 255) & ~(size_t)255;
    char* p = ws + off;
    off += bytes;
    return p;
  };
  float* xc   = (float*)alloc((size_t)3 * TB * 65536 * 4);
  float* offb = (float*)alloc((size_t)50 * N1 * 4);
  float* modb = (float*)alloc((size_t)25 * N1 * 4);
  unsigned short* cols = (unsigned short*)alloc((size_t)1152 * N2 * 2 > (size_t)96 * N1 * 2
                                                ? (size_t)1152 * N2 * 2 : (size_t)96 * N1 * 2);
  unsigned short* wp  = (unsigned short*)alloc((size_t)128 * 1152 * 2);
  unsigned short* wpa = (unsigned short*)alloc((size_t)128 * 1152 * 2);
  unsigned short* wpb = (unsigned short*)alloc((size_t)128 * 1152 * 2);
  float* bufA = (float*)alloc((size_t)32 * N1 * 4);
  float* h1   = (float*)alloc((size_t)32 * N1 * 4);
  float* h2   = (float*)alloc((size_t)64 * N2 * 4);
  float* h3a  = (float*)alloc((size_t)128 * N2 * 4);
  float* h3b  = (float*)alloc((size_t)128 * N2 * 4);
  float* ubuf = (float*)alloc((size_t)128 * N2 * 4);
  float* meanb = (float*)alloc(256 * 4);
  float* rstdb = (float*)alloc(256 * 4);

  L1D(k_nchw_to_cbhw, (long long)TB * 3 * 65536, x_in, xc, 3, 65536);

  // d1: 3->32, k5 s2 p2 -> 128x128
  deform_layer(xc, 3, 256, 256, 32, 5, 2, 2, d1w, d1b, d1wo, d1bo, d1wm, d1bm,
               g1, bt1, offb, modb, cols, wp, bufA, meanb, rstdb, h1, stream);
  // d2: 32->64, k3 s2 p1 -> 64x64
  deform_layer(h1, 32, 128, 128, 64, 3, 2, 1, d2w, d2b, d2wo, d2bo, d2wm, d2bm,
               g2, bt2, offb, modb, cols, wp, bufA, meanb, rstdb, h2, stream);
  // d3: 64->128, k3 s1 p1 -> 64x64
  deform_layer(h2, 64, 64, 64, 128, 3, 1, 1, d3w, d3b, d3wo, d3bo, d3wm, d3bm,
               g3, bt3, offb, modb, cols, wp, bufA, meanb, rstdb, h3a, stream);

  // residual block (shared weights) x4 : u = relu(bn(conv_a(t))); u = bn(conv_b(u)); t = t+u
  const int rK = 128 * 9;  // 1152, multiple of 32
  {
    int MpadT = 128;
    L1D(k_pack_w, (long long)MpadT * rK, rwa, wpa, 128, rK, MpadT, rK);
    L1D(k_pack_w, (long long)MpadT * rK, rwb, wpb, 128, rK, MpadT, rK);
  }
  float* cur = h3a; float* nxt = h3b;
  for (int it = 0; it < 4; ++it) {
    L1D(k_im2col_t, (long long)(rK >> 5) * N2, cur, cols, 128, 64, 64, 3, 1, 1, 64, 64, rK, rK);
    dim3 ga((unsigned)(N2 / GBN), 2u);
    k_wmma_gemm<<<ga, 256, 0, stream>>>(wpa, cols, rba, bufA, 128, (int)N2, rK);
    k_bn_stats<<<128, 256, 0, stream>>>(bufA, meanb, rstdb, N2);
    L1D(k_bn_apply, (long long)128 * N2, bufA, meanb, rstdb, rga, rbta,
        (const float*)nullptr, ubuf, N2, 128, 1);
    L1D(k_im2col_t, (long long)(rK >> 5) * N2, ubuf, cols, 128, 64, 64, 3, 1, 1, 64, 64, rK, rK);
    k_wmma_gemm<<<ga, 256, 0, stream>>>(wpb, cols, rbb, bufA, 128, (int)N2, rK);
    k_bn_stats<<<128, 256, 0, stream>>>(bufA, meanb, rstdb, N2);
    L1D(k_bn_apply, (long long)128 * N2, bufA, meanb, rstdb, rgb, rbtb,
        (const float*)cur, nxt, N2, 128, 0);
    float* tmp = cur; cur = nxt; nxt = tmp;
  }

  // d4: 128->64, k3 s1 p1
  deform_layer(cur, 128, 64, 64, 64, 3, 1, 1, d4w, d4b, d4wo, d4bo, d4wm, d4bm,
               g4, bt4, offb, modb, cols, wp, bufA, meanb, rstdb, ubuf, stream);
  // d5: 64->1, k1 s1 p0 -> final [16,1,64,64] straight to d_out (CBHW==NCHW for C=1)
  deform_layer(ubuf, 64, 64, 64, 1, 1, 1, 0, d5w, d5b, d5wo, d5bo, d5wm, d5bm,
               g5, bt5, offb, modb, cols, wp, bufA, meanb, rstdb, (float*)d_out, stream);
}
